// HierarchicalExpertRouter_77094662963657
// MI455X (gfx1250) — matrix operand order repaired and checked by reading, not compiled, not run on hardware
//
#include <hip/hip_runtime.h>

// HierarchicalExpertRouter on MI455X (gfx1250): bf16 WMMA path.
// B=8192, D=1024, L=3 levels, E=8 experts, RH=128, EH=256.
// Compute-bound (~213 GFLOP vs ~30us of HBM traffic at 23.3 TB/s) -> all big
// GEMMs on v_wmma_f32_16x16x32_bf16, fp32 accumulate. Weights pre-transposed
// once per call so both A and B LDS tiles are straight coalesced copies.

#define BATCH 8192
#define DIM   1024
#define LVLS  3
#define NEXP  8
#define RHID  128
#define EHID  256
#define KC    (NEXP * EHID)   // 2048 = flattened (expert, hidden) contraction dim

// GEMM tiling: 128x128 block tile, K-step 32, 256 threads = 8 wave32s.
#define BM 128
#define BN 128
#define BK 32
#define LDSTR 40              // padded LDS row stride (bf16 elems): 80B = 20 banks -> conflict free

typedef __attribute__((ext_vector_type(16))) __bf16 v16bf;
typedef __attribute__((ext_vector_type(8)))  float  v8f;

union FragAB { v16bf v; uint4 u[2]; };

// ---------------------------------------------------------------- f32 -> bf16 (same layout)
__global__ __launch_bounds__(256) void cvt_f32_bf16(const float* __restrict__ in,
                                                    __bf16* __restrict__ out, int n) {
  int i = (blockIdx.x * 256 + threadIdx.x) * 4;
  if (i + 3 < n) {
    float4 f = *(const float4*)(in + i);
    union { __bf16 b[4]; uint2 u; } o;
    o.b[0] = (__bf16)f.x; o.b[1] = (__bf16)f.y;
    o.b[2] = (__bf16)f.z; o.b[3] = (__bf16)f.w;
    *(uint2*)(out + i) = o.u;
  }
}

// ---------------------------------------------------------------- f32 [R][C] -> bf16 [C][R]
// One-time weight transpose so GEMM B-tiles are n-major (k contiguous per row).
// 32x32 tile per block, 256 threads, LDS-staged (33-float padded rows).
__global__ __launch_bounds__(256) void cvt_transpose_bf16(
    const float* __restrict__ in, __bf16* __restrict__ out, int R, int C) {
  __shared__ float tile[32][33];
  const int t  = threadIdx.x;
  const int c0 = blockIdx.x * 32, r0 = blockIdx.y * 32;
  const float* inm = in + (size_t)blockIdx.z * R * C;
  __bf16*     outm = out + (size_t)blockIdx.z * R * C;
  {
    int r = t >> 3, c4 = (t & 7) * 4;
    float4 f = *(const float4*)(inm + (size_t)(r0 + r) * C + c0 + c4);
    tile[r][c4]     = f.x; tile[r][c4 + 1] = f.y;
    tile[r][c4 + 2] = f.z; tile[r][c4 + 3] = f.w;
  }
  __syncthreads();
  {
    int c = t >> 3, r4 = (t & 7) * 4;
    union { __bf16 b[4]; uint2 u; } o;
    #pragma unroll
    for (int j = 0; j < 4; ++j) o.b[j] = (__bf16)tile[r4 + j][c];
    *(uint2*)(outm + (size_t)(c0 + c) * R + r0 + r4) = o.u;
  }
}

// ---------------------------------------------------------------- router
// 8 rows per block, 128 threads. rh = relu(x@rw1+rb1); w = softmax(rh@rw2+rb2).
__global__ __launch_bounds__(128) void router_kernel(
    const float* __restrict__ cur, const float* __restrict__ rw1,
    const float* __restrict__ rb1, const float* __restrict__ rw2,
    const float* __restrict__ rb2, float* __restrict__ wrt) {
  __shared__ float xs[8][DIM];
  __shared__ float rh[8][RHID];
  __shared__ float lg[8][NEXP];
  const int tid = threadIdx.x;
  const int r0  = blockIdx.x * 8;

  for (int i = 0; i < 64; ++i) {               // stage 8 input rows (32 KB LDS)
    int c = tid + i * 128;
    xs[c >> 10][c & (DIM - 1)] = cur[(size_t)(r0 + (c >> 10)) * DIM + (c & (DIM - 1))];
  }
  __syncthreads();

  float acc[8];
  const float b1 = rb1[tid];
  for (int r = 0; r < 8; ++r) acc[r] = b1;
  for (int k = 0; k < DIM; ++k) {
    float w = rw1[k * RHID + tid];             // coalesced across threads
    #pragma unroll
    for (int r = 0; r < 8; ++r) acc[r] = fmaf(xs[r][k], w, acc[r]);
  }
  for (int r = 0; r < 8; ++r) rh[r][tid] = fmaxf(acc[r], 0.f);
  __syncthreads();

  if (tid < 64) {                              // one thread per (row, expert) logit
    int r = tid >> 3, e = tid & 7;
    float s = rb2[e];
    for (int j = 0; j < RHID; ++j) s = fmaf(rh[r][j], rw2[j * NEXP + e], s);
    lg[r][e] = s;
  }
  __syncthreads();
  if (tid < 8) {                               // per-row softmax over 8 experts
    float m = lg[tid][0];
    for (int e = 1; e < 8; ++e) m = fmaxf(m, lg[tid][e]);
    float ex[8], s = 0.f;
    for (int e = 0; e < 8; ++e) { ex[e] = __expf(lg[tid][e] - m); s += ex[e]; }
    float inv = 1.f / s;
    for (int e = 0; e < 8; ++e) wrt[(size_t)(r0 + tid) * NEXP + e] = ex[e] * inv;
  }
}

// ---------------------------------------------------------------- shared GEMM body
// A: [M][K] row-major bf16 (astride = K). B: n-major bf16, row n holds K
// contiguous (bstride = K). Both tiles stage as straight uint4 copies into
// LDS [row][40]; fragments per ISA 7.12.2:
//   A frag lane L: row m+(L&15), two 16B chunks at k = (L>>4)*8 and 16+(L>>4)*8
//   B frag lane L: row n+(L&15), two 16B chunks at k = (L>>4)*8 and 16+(L>>4)*8
__device__ __forceinline__ void gemm_tile_bf16(
    const __bf16* __restrict__ asrc, int astride,
    const __bf16* __restrict__ bsrc, int bstride, int K,
    __bf16* As, __bf16* Bs, int tid, v8f acc[2][4]) {
  const int wv = tid >> 5, lane = tid & 31;
  const int wm = wv & 3,  wn = wv >> 2;        // 4x2 wave grid over 128x128 tile
  const int lrow = lane & 15, lhalf = lane >> 4;
  const int crow = tid >> 2, ckc = (tid & 3) * 8;   // loader chunk (2 chunks/thread)

  for (int k0 = 0; k0 < K; k0 += BK) {
    #pragma unroll
    for (int i = 0; i < 2; ++i) {              // A tile: 128x32 coalesced copy
      int row = crow + i * 64;
      *(uint4*)(As + row * LDSTR + ckc) =
          *(const uint4*)(asrc + (size_t)row * astride + k0 + ckc);
    }
    #pragma unroll
    for (int i = 0; i < 2; ++i) {              // B tile: 128x32 coalesced copy (pre-transposed)
      int row = crow + i * 64;
      *(uint4*)(Bs + row * LDSTR + ckc) =
          *(const uint4*)(bsrc + (size_t)row * bstride + k0 + ckc);
    }
    if (k0 + BK < K) {                         // prefetch next k-step tiles into cache
      __builtin_prefetch(asrc + (size_t)crow * astride + k0 + BK + ckc, 0, 0);
      __builtin_prefetch(bsrc + (size_t)crow * bstride + k0 + BK + ckc, 0, 0);
    }
    __syncthreads();
    FragAB a[2], b[4];
    #pragma unroll
    for (int mi = 0; mi < 2; ++mi) {
      const __bf16* p = As + (wm * 32 + mi * 16 + lrow) * LDSTR + lhalf * 8;
      a[mi].u[0] = *(const uint4*)p;
      a[mi].u[1] = *(const uint4*)(p + 16);
    }
    #pragma unroll
    for (int ni = 0; ni < 4; ++ni) {
      const __bf16* p = Bs + (wn * 64 + ni * 16 + lrow) * LDSTR + lhalf * 8;
      b[ni].u[0] = *(const uint4*)p;
      b[ni].u[1] = *(const uint4*)(p + 16);
    }
    #pragma unroll
    for (int mi = 0; mi < 2; ++mi)
      #pragma unroll
      for (int ni = 0; ni < 4; ++ni)
        acc[mi][ni] = __builtin_amdgcn_wmma_f32_16x16x32_bf16(
            false, a[mi].v, false, b[ni].v, (short)0, acc[mi][ni], false, false);
    __syncthreads();
  }
}

// ---------------------------------------------------------------- expert GEMM1
// H[b, e*256+h] = bf16( w[b,e] * relu( X[b,:] @ W1[e][:,h] + b1[e,h] ) )
__global__ __launch_bounds__(256) void expert1_kernel(
    const __bf16* __restrict__ xbf,   // [B][DIM]
    const __bf16* __restrict__ w1t,   // [E][EHID][DIM] transposed (level offset applied)
    const float*  __restrict__ eb1,   // [E][EHID]
    const float*  __restrict__ wrt,   // [B][E]
    __bf16* __restrict__ hout) {      // [B][KC]
  __shared__ __attribute__((aligned(16))) __bf16 As[BM * LDSTR];
  __shared__ __attribute__((aligned(16))) __bf16 Bs[BN * LDSTR];
  const int tid   = threadIdx.x;
  const int m0    = blockIdx.x * BM;
  const int gn0   = blockIdx.y * BN;           // flattened (e,h) column base
  const int e     = gn0 >> 8;                  // EHID=256 -> tile within one expert
  const int n0loc = gn0 & (EHID - 1);

  v8f acc[2][4] = {};
  gemm_tile_bf16(xbf + (size_t)m0 * DIM, DIM,
                 w1t + ((size_t)e * EHID + n0loc) * DIM, DIM, DIM,
                 As, Bs, tid, acc);

  const int wv = tid >> 5, lane = tid & 31;
  const int wm = wv & 3,  wn = wv >> 2;
  const int lrow = lane & 15, lhalf = lane >> 4;
  #pragma unroll
  for (int mi = 0; mi < 2; ++mi) {
    int rbase = m0 + wm * 32 + mi * 16 + lhalf * 8;
    float wvv[8];
    #pragma unroll
    for (int r = 0; r < 8; ++r) wvv[r] = wrt[(size_t)(rbase + r) * NEXP + e];
    #pragma unroll
    for (int ni = 0; ni < 4; ++ni) {
      int n = gn0 + wn * 64 + ni * 16 + lrow;
      float bias = eb1[e * EHID + (n & (EHID - 1))];
      #pragma unroll
      for (int r = 0; r < 8; ++r) {
        float v = fmaxf(acc[mi][ni][r] + bias, 0.f) * wvv[r];
        hout[(size_t)(rbase + r) * KC + n] = (__bf16)v;
      }
    }
  }
}

// ---------------------------------------------------------------- expert GEMM2
// out[b,d] = H[b,:] @ W2cat[:,d] + sum_e w[b,e]*b2[e,d]
__global__ __launch_bounds__(256) void expert2_kernel(
    const __bf16* __restrict__ hbf,   // [B][KC]
    const __bf16* __restrict__ w2t,   // [DIM][KC] transposed (level offset applied)
    const float*  __restrict__ eb2,   // [E][DIM]
    const float*  __restrict__ wrt,   // [B][E]
    float* __restrict__ outp) {       // [B][DIM]
  __shared__ __attribute__((aligned(16))) __bf16 As[BM * LDSTR];
  __shared__ __attribute__((aligned(16))) __bf16 Bs[BN * LDSTR];
  const int tid = threadIdx.x;
  const int m0  = blockIdx.x * BM;
  const int n0  = blockIdx.y * BN;

  v8f acc[2][4] = {};
  gemm_tile_bf16(hbf + (size_t)m0 * KC, KC,
                 w2t + (size_t)n0 * KC, KC, KC,
                 As, Bs, tid, acc);

  const int wv = tid >> 5, lane = tid & 31;
  const int wm = wv & 3,  wn = wv >> 2;
  const int lrow = lane & 15, lhalf = lane >> 4;
  #pragma unroll
  for (int mi = 0; mi < 2; ++mi) {
    int rbase = m0 + wm * 32 + mi * 16 + lhalf * 8;
    #pragma unroll
    for (int ni = 0; ni < 4; ++ni) {
      int d = n0 + wn * 64 + ni * 16 + lrow;
      float be[8];
      #pragma unroll
      for (int e2 = 0; e2 < NEXP; ++e2) be[e2] = eb2[e2 * DIM + d];
      #pragma unroll
      for (int r = 0; r < 8; ++r) {
        const float* wp = wrt + (size_t)(rbase + r) * NEXP;
        float v = acc[mi][ni][r];
        #pragma unroll
        for (int e2 = 0; e2 < NEXP; ++e2) v = fmaf(wp[e2], be[e2], v);
        outp[(size_t)(rbase + r) * DIM + d] = v;
      }
    }
  }
}

// ---------------------------------------------------------------- launch
extern "C" void kernel_launch(void* const* d_in, const int* in_sizes, int n_in,
                              void* d_out, int out_size, void* d_ws, size_t ws_size,
                              hipStream_t stream) {
  const float* x   = (const float*)d_in[0];
  const float* rw1 = (const float*)d_in[1];
  const float* rb1 = (const float*)d_in[2];
  const float* rw2 = (const float*)d_in[3];
  const float* rb2 = (const float*)d_in[4];
  const float* ew1 = (const float*)d_in[5];
  const float* eb1 = (const float*)d_in[6];
  const float* ew2 = (const float*)d_in[7];
  const float* eb2 = (const float*)d_in[8];

  char* ws = (char*)d_ws;
  size_t off = 0;
  auto alloc = [&](size_t bytes) -> void* {
    void* p = ws + off;
    off = (off + bytes + 255) & ~(size_t)255;
    return p;
  };
  __bf16* xbf  = (__bf16*)alloc((size_t)BATCH * DIM * 2);              // 16 MB
  __bf16* w1t  = (__bf16*)alloc((size_t)LVLS * NEXP * EHID * DIM * 2); // 12.6 MB (transposed)
  __bf16* w2t  = (__bf16*)alloc((size_t)LVLS * DIM * KC * 2);          // 12.6 MB (transposed)
  float*  wrt  = (float*) alloc((size_t)BATCH * NEXP * 4);             // 256 KB
  __bf16* hbf  = (__bf16*)alloc((size_t)BATCH * KC * 2);               // 32 MB
  float*  cur  = (float*) alloc((size_t)BATCH * DIM * 4);              // 32 MB

  // One-time weight convert+transpose:
  //   ew1 [L*E][DIM][EHID] -> w1t [L*E][EHID][DIM]
  //   ew2 [L][KC][DIM]     -> w2t [L][DIM][KC]   (rows = flattened (e,h))
  cvt_transpose_bf16<<<dim3(EHID / 32, DIM / 32, LVLS * NEXP), 256, 0, stream>>>(
      ew1, w1t, DIM, EHID);
  cvt_transpose_bf16<<<dim3(DIM / 32, KC / 32, LVLS), 256, 0, stream>>>(
      ew2, w2t, KC, DIM);

  const float* lvlin = x;
  for (int l = 0; l < LVLS; ++l) {
    const int nx = BATCH * DIM;
    cvt_f32_bf16<<<nx / 1024, 256, 0, stream>>>(lvlin, xbf, nx);
    router_kernel<<<BATCH / 8, 128, 0, stream>>>(
        lvlin, rw1 + (size_t)l * DIM * RHID, rb1 + (size_t)l * RHID,
        rw2 + (size_t)l * RHID * NEXP, rb2 + (size_t)l * NEXP, wrt);
    expert1_kernel<<<dim3(BATCH / BM, KC / BN), 256, 0, stream>>>(
        xbf, w1t + (size_t)l * NEXP * EHID * DIM,
        eb1 + (size_t)l * NEXP * EHID, wrt, hbf);
    float* lvlout = (l == LVLS - 1) ? (float*)d_out : cur;
    expert2_kernel<<<dim3(BATCH / BM, DIM / BN), 256, 0, stream>>>(
        hbf, w2t + (size_t)l * DIM * KC,
        eb2 + (size_t)l * NEXP * DIM, wrt, lvlout);
    lvlin = cur;
  }
}
